// Copa_59184649339497
// MI455X (gfx1250) — matrix operand order repaired and checked by reading, not compiled, not run on hardware
//
#include <hip/hip_runtime.h>
#include <hip/hip_bf16.h>

typedef __attribute__((ext_vector_type(16))) _Float16 v16h;
typedef __attribute__((ext_vector_type(8)))  _Float16 v8h;
typedef __attribute__((ext_vector_type(8)))  float    v8f;
typedef __attribute__((ext_vector_type(4)))  float    v4f;

#define WAVES_PER_BLOCK 8

// DPP16 butterfly add: swap with partner via dpp control, add. No LDS traffic.
__device__ __forceinline__ float dpp_swap_add(float v, const int ctrl) {
  int src = __float_as_int(v);
  int prm;
  switch (ctrl) {  // dpp_ctrl must be an immediate
    case 0: prm = __builtin_amdgcn_update_dpp(0, src, 0x0B1, 0xf, 0xf, true); break; // quad_perm(1,0,3,2) : xor 1
    case 1: prm = __builtin_amdgcn_update_dpp(0, src, 0x04E, 0xf, 0xf, true); break; // quad_perm(2,3,0,1) : xor 2
    case 2: prm = __builtin_amdgcn_update_dpp(0, src, 0x141, 0xf, 0xf, true); break; // row_half_mirror
    default:prm = __builtin_amdgcn_update_dpp(0, src, 0x140, 0xf, 0xf, true); break; // row_mirror
  }
  return v + __int_as_float(prm);
}

// Sum over each 16-lane half-wave (rows of the D-tile lane mapping).
// quad_perm steps make quads uniform, then mirrors are xor-equivalent.
__device__ __forceinline__ float halfwave_sum(float v) {
  v = dpp_swap_add(v, 0);
  v = dpp_swap_add(v, 1);
  v = dpp_swap_add(v, 2);
  v = dpp_swap_add(v, 3);
  return v;
}

// One wave handles 32 batches (two 16x16 WMMA tiles).
//   A1(x, 16x24 -> f16 16x32)  --wmma--> Dk, Dq   (k/q projections)
//   kq = (Dk+bk)*(Dq+bq)       (elementwise, C/D layout)
//   LDS transpose (D layout -> A layout, f16)
//   A2(kq) --wmma--> x5 channels 0-15 (B5a) and 16-23 (B5b)  [W54 = w5@w4]
//   LayerNorm over 24 ch via DPP16 butterflies, sigmoid(relu), gate by x.
__global__ __launch_bounds__(WAVES_PER_BLOCK * 32) void copa_wmma_kernel(
    const float* __restrict__ x,
    const float* __restrict__ wk, const float* __restrict__ bk,
    const float* __restrict__ wq, const float* __restrict__ bq,
    const float* __restrict__ w4, const float* __restrict__ b4,
    const float* __restrict__ w5, const float* __restrict__ b5,
    const float* __restrict__ gamma, const float* __restrict__ beta,
    float* __restrict__ out, int B)
{
  __shared__ __align__(64) _Float16 lds_kq[WAVES_PER_BLOCK][16 * 16];

  const int wave = threadIdx.x >> 5;
  const int lane = threadIdx.x & 31;
  const int n    = lane & 15;   // channel index within a 16-wide D tile
  const int hi   = lane >> 4;   // half-wave select

  // ---------------- one-time weight staging (per wave) ----------------
  // B layout (f16 32x16): lane l holds column N=l%16, component s = K (8h+s | 16+8h+s-8).
  v16h Bk = {}, Bq = {}, B5a = {}, B5b = {};
  if (n < 12) {
    #pragma unroll
    for (int c = 0; c < 16; ++c) {
      const int kk = (c < 8) ? (hi * 8 + c) : (16 + hi * 8 + c - 8);
      if (kk < 24) {
        Bk[c] = (_Float16)wk[n * 24 + kk];   // Bk[K=kk][N=n] = wk[n][kk]
        Bq[c] = (_Float16)wq[n * 24 + kk];
      }
    }
  }
  {
    // W54 = w5 @ w4 : [24,12]; split into out-channels 0-15 / 16-23
    #pragma unroll
    for (int c = 0; c < 8; ++c) {
      const int kk = hi * 8 + c;
      if (kk < 12) {
        float acc = 0.f;
        #pragma unroll
        for (int j = 0; j < 6; ++j) acc += w5[n * 6 + j] * w4[j * 12 + kk];
        B5a[c] = (_Float16)acc;
        if (n < 8) {
          float acc2 = 0.f;
          #pragma unroll
          for (int j = 0; j < 6; ++j) acc2 += w5[(16 + n) * 6 + j] * w4[j * 12 + kk];
          B5b[c] = (_Float16)acc2;
        }
      }
    }
  }
  // fused biases b54 = w5@b4 + b5 for this lane's channels n and 16+n
  float b54a = b5[n];
  #pragma unroll
  for (int j = 0; j < 6; ++j) b54a += w5[n * 6 + j] * b4[j];
  float b54b = 0.f;
  if (n < 8) {
    b54b = b5[16 + n];
    #pragma unroll
    for (int j = 0; j < 6; ++j) b54b += w5[(16 + n) * 6 + j] * b4[j];
  }
  const float bkv = (n < 12) ? bk[n] : 0.f;
  const float bqv = (n < 12) ? bq[n] : 0.f;
  const float gmA = gamma[n], btA = beta[n];
  const float gmB = (n < 8) ? gamma[16 + n] : 0.f;
  const float btB = (n < 8) ? beta[16 + n]  : 0.f;

  const long tileBase = ((long)blockIdx.x * WAVES_PER_BLOCK + wave) * 32;
  if (tileBase >= B) return;
  _Float16* kqbuf = lds_kq[wave];
  const v8f zero8 = {};

  #pragma unroll
  for (int s = 0; s < 2; ++s) {
    const long mbase = tileBase + s * 16;

    // speculative prefetch of next subtile's activations
    __builtin_prefetch((const void*)(x + (mbase + 16 + n) * 24), 0, 3);

    // ---- A1: x tile, A layout. lane<16: K 0-7 & 16-23; lane>=16: K 8-15; K24-31 = 0
    const v4f* xp = (const v4f*)(x + (mbase + n) * 24);  // 96B rows -> 16B aligned
    v4f q0 = xp[hi ? 2 : 0];
    v4f q1 = xp[hi ? 3 : 1];
    v4f q2 = xp[4], q3 = xp[5];
    if (hi) {
      v4f zz = {0.f, 0.f, 0.f, 0.f};
      q2 = zz; q3 = zz;
    }
    v16h A;
    #pragma unroll
    for (int i = 0; i < 4; ++i) {
      A[i]      = (_Float16)q0[i];
      A[4 + i]  = (_Float16)q1[i];
      A[8 + i]  = (_Float16)q2[i];
      A[12 + i] = (_Float16)q3[i];
    }

    // ---- k/q projections: D = A x B + 0
    v8f Dk = __builtin_amdgcn_wmma_f32_16x16x32_f16(false, A, false, Bk,
                                                    (short)0, zero8, false, false);
    v8f Dq = __builtin_amdgcn_wmma_f32_16x16x32_f16(false, A, false, Bq,
                                                    (short)0, zero8, false, false);

    // ---- kq elementwise in D layout, staged to LDS (f16) for the transpose
    #pragma unroll
    for (int r = 0; r < 8; ++r) {
      float kq = (Dk[r] + bkv) * (Dq[r] + bqv);
      kqbuf[(hi * 8 + r) * 16 + n] = (_Float16)kq;  // row = batch, col = channel
    }
    asm volatile("s_wait_dscnt 0" ::: "memory");

    // ---- A2: read back in A layout (lane<16: ch 0-7; lane>=16: ch 8-15; K16-31 = 0)
    v16h A2 = {};
    {
      v8h lo = *(const v8h*)(kqbuf + n * 16 + hi * 8);  // 16B-aligned ds_load_b128
      #pragma unroll
      for (int i = 0; i < 8; ++i) A2[i] = lo[i];
    }
    asm volatile("" ::: "memory");  // keep this load ahead of next subtile's stores

    // ---- fused 12->24 projection (W54), split over two 16-wide N tiles
    v8f D5a = __builtin_amdgcn_wmma_f32_16x16x32_f16(false, A2, false, B5a,
                                                     (short)0, zero8, false, false);
    v8f D5b = __builtin_amdgcn_wmma_f32_16x16x32_f16(false, A2, false, B5b,
                                                     (short)0, zero8, false, false);

    // ---- LayerNorm(24) + sigmoid(relu) + gate, per batch row
    #pragma unroll
    for (int r = 0; r < 8; ++r) {
      float a  = D5a[r] + b54a;                       // channel n
      float bb = (n < 8) ? (D5b[r] + b54b) : 0.f;     // channel 16+n (else 0)

      float t  = halfwave_sum(a + bb);                // sum over 24 channels (DPP)
      float mu = t * (1.f / 24.f);

      float da = a - mu;
      float db = bb - mu;
      float vv = halfwave_sum(da * da + ((n < 8) ? db * db : 0.f));
      float inv = rsqrtf(vv * (1.f / 24.f) + 1e-5f);

      float ya = da * inv * gmA + btA;
      float yb = db * inv * gmB + btB;
      float ga = __builtin_amdgcn_rcpf(1.f + __expf(-fmaxf(ya, 0.f)));
      float gb = __builtin_amdgcn_rcpf(1.f + __expf(-fmaxf(yb, 0.f)));

      const long gm = mbase + hi * 8 + r;
      const float* xrow = x + gm * 24;                // WGP$-hot re-read for gate
      out[gm * 24 + n] = xrow[n] * ga;
      if (n < 8) out[gm * 24 + 16 + n] = xrow[16 + n] * gb;
    }
  }
}

extern "C" void kernel_launch(void* const* d_in, const int* in_sizes, int n_in,
                              void* d_out, int out_size, void* d_ws, size_t ws_size,
                              hipStream_t stream) {
  const float* x     = (const float*)d_in[0];
  const float* wk    = (const float*)d_in[1];
  const float* bk    = (const float*)d_in[2];
  const float* wq    = (const float*)d_in[3];
  const float* bq    = (const float*)d_in[4];
  // d_in[5] = wv, d_in[6] = bv: dead code (softmax over a singleton axis == 1)
  const float* w4    = (const float*)d_in[7];
  const float* b4    = (const float*)d_in[8];
  const float* w5    = (const float*)d_in[9];
  const float* b5    = (const float*)d_in[10];
  const float* gamma = (const float*)d_in[11];
  const float* beta  = (const float*)d_in[12];
  float* out = (float*)d_out;

  const int B = in_sizes[0] / 24;                       // 1,048,576
  const int tiles  = (B + 31) / 32;                     // 32 batches per wave
  const int blocks = (tiles + WAVES_PER_BLOCK - 1) / WAVES_PER_BLOCK;

  copa_wmma_kernel<<<blocks, WAVES_PER_BLOCK * 32, 0, stream>>>(
      x, wk, bk, wq, bq, w4, b4, w5, b5, gamma, beta, out, B);
}
